// SABlock_11742440587448
// MI455X (gfx1250) — matrix operand order repaired and checked by reading, not compiled, not run on hardware
//
#include <hip/hip_runtime.h>

#define B_    4
#define N_    1025
#define C_    768
#define H_    12
#define D_    64
#define NP    1056      // padded token count for q/k/vt (33*32)
#define NROWP 1152      // padded row space for xh/ctx (9 blocks * 128 rows)
#define TOPK  103
#define SCALE 0.125f
#define NEGBIG (-3.0e38f)

typedef __attribute__((ext_vector_type(16))) _Float16 v16h;
typedef __attribute__((ext_vector_type(8)))  _Float16 v8h;
typedef __attribute__((ext_vector_type(8)))  float    v8f;

static constexpr size_t OFF_ENHI = (size_t)B_ * N_ * C_;                    // enhance_index
static constexpr size_t OFF_ENHS = OFF_ENHI + (size_t)B_ * TOPK * C_;       // enh_idx
static constexpr size_t OFF_FUSI = OFF_ENHS + (size_t)B_ * TOPK;            // fuse_index
static constexpr size_t OFF_FUSS = OFF_FUSI + (size_t)B_ * TOPK * C_;       // fuse_idx
static constexpr size_t OFF_CLS  = OFF_FUSS + (size_t)B_ * TOPK;            // cls_attn

__device__ __forceinline__ v8f wmma16(v16h a, v16h b, v8f c) {
  return __builtin_amdgcn_wmma_f32_16x16x32_f16(false, a, false, b, (short)0, c,
                                                false, false);
}

// A fragment (16x32 f16). Lane (m = lane&15, half = lane>>4) holds row m,
// k in [kc+half*16, +16) (contiguous, matching the B fragments).
__device__ __forceinline__ v16h a_frag_f16(const _Float16* __restrict__ A, int lda,
                                           int row, int kc, int half) {
  const _Float16* p = A + (size_t)row * lda + kc + half * 16;
  v8h lo = *(const v8h*)p;
  v8h hi = *(const v8h*)(p + 8);
  v16h f;
#pragma unroll
  for (int i = 0; i < 8; ++i) { f[i] = lo[i]; f[8 + i] = hi[i]; }
  return f;
}

// B fragment (32x16 f16) for out = A @ W^T with W row-major f16 [cols][ldk]:
// lane (n, half) holds column n, k in [kc+half*16, +16) -- one 32B load.
__device__ __forceinline__ v16h b_frag_f16(const _Float16* __restrict__ W, int ldk,
                                           int col, int kc, int half) {
  return *(const v16h*)(W + (size_t)col * ldk + kc + half * 16);
}

// ---------------------------------------------------------------------------
// f16 zero fill (n multiple of 8)
// ---------------------------------------------------------------------------
__global__ void __launch_bounds__(256) zero_f16(_Float16* __restrict__ dst, int n) {
  int i = (blockIdx.x * 256 + threadIdx.x) * 8;
  if (i + 8 <= n) {
    v8h z;
#pragma unroll
    for (int j = 0; j < 8; ++j) z[j] = (_Float16)0.0f;
    *(v8h*)(dst + i) = z;
  }
}

// ---------------------------------------------------------------------------
// fp32 -> f16 bulk conversion, contiguous (n multiple of 8)
// ---------------------------------------------------------------------------
__global__ void __launch_bounds__(256) cvt_f16(const float* __restrict__ src,
                                               _Float16* __restrict__ dst, int n) {
  int i = (blockIdx.x * 256 + threadIdx.x) * 8;
  if (i + 8 <= n) {
    v8h o;
#pragma unroll
    for (int j = 0; j < 8; ++j) o[j] = (_Float16)src[i + j];
    *(v8h*)(dst + i) = o;
  }
}

// ---------------------------------------------------------------------------
// x [B][N_][C_] fp32 -> xh [B][NROWP][C_] f16 (valid rows only)
// ---------------------------------------------------------------------------
__global__ void __launch_bounds__(256) cvt_x(const float* __restrict__ src,
                                             _Float16* __restrict__ dst) {
  size_t i = ((size_t)blockIdx.x * 256 + threadIdx.x) * 8;
  const size_t per_b = (size_t)N_ * C_;
  if (i + 8 <= (size_t)B_ * per_b) {
    int b = (int)(i / per_b);
    size_t rem = i - (size_t)b * per_b;
    v8h o;
#pragma unroll
    for (int j = 0; j < 8; ++j) o[j] = (_Float16)src[i + j];
    *(v8h*)(dst + (size_t)b * NROWP * C_ + rem) = o;
  }
}

// ---------------------------------------------------------------------------
// QKV projection: qkv = xh @ wqkvh^T, scattered into q/k (row-major
// [b,h,tok,d]) and vt (transposed [b,h,d,tok]). All-f16 fragments.
// ---------------------------------------------------------------------------
__global__ void __launch_bounds__(256) qkv_kernel(
    const _Float16* __restrict__ xh, const _Float16* __restrict__ wqkvh,
    _Float16* __restrict__ q, _Float16* __restrict__ k, _Float16* __restrict__ vt) {
  const int tid  = threadIdx.x;
  const int lane = tid & 31;
  const int wv   = tid >> 5;
  const int half = lane >> 4;
  const int n    = lane & 15;
  const int b    = blockIdx.z;
  const int row0 = blockIdx.x * 128 + wv * 16;
  const int col0 = blockIdx.y * 64;
  const _Float16* xb = xh + (size_t)b * NROWP * C_;
  const int arow = row0 + n;

  v8f acc[4];
#pragma unroll
  for (int cg = 0; cg < 4; ++cg)
#pragma unroll
    for (int r = 0; r < 8; ++r) acc[cg][r] = 0.0f;

  for (int kc = 0; kc < C_; kc += 64) {
    v16h af0 = a_frag_f16(xb, C_, arow, kc,      half);
    v16h af1 = a_frag_f16(xb, C_, arow, kc + 32, half);
#pragma unroll
    for (int cg = 0; cg < 4; ++cg) {
      const int col = col0 + cg * 16 + n;
      v16h bf0 = b_frag_f16(wqkvh, C_, col, kc,      half);
      v16h bf1 = b_frag_f16(wqkvh, C_, col, kc + 32, half);
      acc[cg] = wmma16(af0, bf0, acc[cg]);
      acc[cg] = wmma16(af1, bf1, acc[cg]);
    }
  }

#pragma unroll
  for (int cg = 0; cg < 4; ++cg) {
    const int j = col0 + cg * 16 + n;
#pragma unroll
    for (int r = 0; r < 8; ++r) {
      const int row = row0 + half * 8 + r;
      if (row >= NP) continue;
      _Float16 v = (_Float16)acc[cg][r];
      if (j < 768) {
        int h = j >> 6, d = j & 63;
        q[(((size_t)b * H_ + h) * NP + row) * D_ + d] = v;
      } else if (j < 1536) {
        int jj = j - 768; int h = jj >> 6, d = jj & 63;
        k[(((size_t)b * H_ + h) * NP + row) * D_ + d] = v;
      } else {
        int jj = j - 1536; int h = jj >> 6, d = jj & 63;
        vt[(((size_t)b * H_ + h) * D_ + d) * NP + row] = v;
      }
    }
  }
}

// ---------------------------------------------------------------------------
// Attention: one block per (b, h, 16-row tile). Full score row in LDS (fp32),
// exact softmax, normalized P stored as f16 in LDS, P@V via WMMA,
// cls-attn accumulation for row tile 0.
// ---------------------------------------------------------------------------
#define SCN 1064
__global__ void __launch_bounds__(128) attn_kernel(
    const _Float16* __restrict__ q, const _Float16* __restrict__ k,
    const _Float16* __restrict__ vt, _Float16* __restrict__ ctx,
    float* __restrict__ cls) {
  __shared__ float    sc[16][SCN];
  __shared__ _Float16 pf[16][SCN];
  __shared__ float    red[16][8];
  __shared__ float    rowstat[16];

  const int tid  = threadIdx.x;
  const int lane = tid & 31;
  const int wv   = tid >> 5;
  const int half = lane >> 4;
  const int n    = lane & 15;
  const int rt = blockIdx.x, h = blockIdx.y, b = blockIdx.z;
  const int row0 = rt * 16;
  const size_t bh = (size_t)b * H_ + h;
  const _Float16* qp = q  + bh * NP * D_;
  const _Float16* kp = k  + bh * NP * D_;
  const _Float16* vp = vt + bh * (size_t)D_ * NP;

  // Q fragments for K-dim = 64 (two 32-chunks)
  v16h qa0 = a_frag_f16(qp, D_, row0 + n, 0,  half);
  v16h qa1 = a_frag_f16(qp, D_, row0 + n, 32, half);

  // Phase 1: scores into LDS (waves stripe the 66 column tiles)
  for (int ct = wv; ct < 66; ct += 4) {
    const int col0 = ct * 16;
    const _Float16* kr = kp + (size_t)(col0 + n) * D_;
    v16h kb0 = *(const v16h*)(kr + half * 16);
    v16h kb1 = *(const v16h*)(kr + 32 + half * 16);
    v8f s;
#pragma unroll
    for (int r = 0; r < 8; ++r) s[r] = 0.0f;
    s = wmma16(qa0, kb0, s);
    s = wmma16(qa1, kb1, s);
#pragma unroll
    for (int r = 0; r < 8; ++r) {
      const int row = row0 + half * 8 + r;
      const int col = col0 + n;
      float v = s[r] * SCALE;
      bool colmask = (col >= N_);
      bool blkmask = (row >= 1) && (col >= 1) && (row < N_) && (col < N_) &&
                     (((row - 1) >> 8) == ((col - 1) >> 8)) && (((row - 1) >> 8) < 4);
      if (colmask || blkmask) v = NEGBIG;
      sc[half * 8 + r][col] = v;
    }
  }
  __syncthreads();

  // Phase 2: softmax (8 threads per row); sc keeps exp(), pf gets normalized f16
  const int rr = tid >> 3, jj = tid & 7;
  float m = NEGBIG;
  for (int c = jj; c < NP; c += 8) m = fmaxf(m, sc[rr][c]);
  red[rr][jj] = m;
  __syncthreads();
  if (jj == 0) {
    float mm = red[rr][0];
#pragma unroll
    for (int i = 1; i < 8; ++i) mm = fmaxf(mm, red[rr][i]);
    rowstat[rr] = mm;
  }
  __syncthreads();
  const float rmax = rowstat[rr];
  float ssum = 0.0f;
  for (int c = jj; c < NP; c += 8) {
    float e = __expf(sc[rr][c] - rmax);
    sc[rr][c] = e;
    ssum += e;
  }
  red[rr][jj] = ssum;
  __syncthreads();
  if (jj == 0) {
    float t = 0.0f;
#pragma unroll
    for (int i = 0; i < 8; ++i) t += red[rr][i];
    rowstat[rr] = t;
  }
  __syncthreads();
  const float rinv = 1.0f / rowstat[rr];
  for (int c = jj; c < NP; c += 8) pf[rr][c] = (_Float16)(sc[rr][c] * rinv);
  __syncthreads();

  // cls attention (row 0 of tile 0), averaged over heads via atomics (fp32 path)
  if (rt == 0) {
    const float rinv0 = (1.0f / H_) / rowstat[0];
    for (int c = 1 + tid; c < N_; c += 128)
      atomicAdd(&cls[(size_t)b * 1024 + (c - 1)], sc[0][c] * rinv0);
  }

  // Phase 3: ctx = P @ V (wave wv owns d-columns [wv*16, +16))
  v8f acc;
#pragma unroll
  for (int r = 0; r < 8; ++r) acc[r] = 0.0f;
  const int d = wv * 16 + n;
  for (int kc = 0; kc < NP; kc += 32) {
    v16h pa;
    {
      const _Float16* p = &pf[n][kc + half * 16];
      v8h lo = *(const v8h*)p;
      v8h hi = *(const v8h*)(p + 8);
#pragma unroll
      for (int i = 0; i < 8; ++i) { pa[i] = lo[i]; pa[8 + i] = hi[i]; }
    }
    v16h vb = *(const v16h*)(vp + (size_t)d * NP + kc + half * 16);
    acc = wmma16(pa, vb, acc);
  }
#pragma unroll
  for (int r = 0; r < 8; ++r) {
    const int row = row0 + half * 8 + r;   // <= 1039 < NROWP
    ctx[((size_t)b * NROWP + row) * C_ + h * D_ + d] = (_Float16)acc[r];
  }
}

// ---------------------------------------------------------------------------
// Output projection: out = ctx @ wouth^T + b_out (fp32 stores)
// ---------------------------------------------------------------------------
__global__ void __launch_bounds__(256) outproj_kernel(
    const _Float16* __restrict__ ctx, const _Float16* __restrict__ wouth,
    const float* __restrict__ bout, float* __restrict__ out) {
  const int tid  = threadIdx.x;
  const int lane = tid & 31;
  const int wv   = tid >> 5;
  const int half = lane >> 4;
  const int n    = lane & 15;
  const int b    = blockIdx.z;
  const int row0 = blockIdx.x * 128 + wv * 16;
  const int col0 = blockIdx.y * 64;
  const _Float16* A = ctx + (size_t)b * NROWP * C_;
  const int arow = row0 + n;

  v8f acc[4];
#pragma unroll
  for (int cg = 0; cg < 4; ++cg)
#pragma unroll
    for (int r = 0; r < 8; ++r) acc[cg][r] = 0.0f;

  for (int kc = 0; kc < C_; kc += 64) {
    v16h af0 = a_frag_f16(A, C_, arow, kc,      half);
    v16h af1 = a_frag_f16(A, C_, arow, kc + 32, half);
#pragma unroll
    for (int cg = 0; cg < 4; ++cg) {
      const int col = col0 + cg * 16 + n;
      v16h bf0 = b_frag_f16(wouth, C_, col, kc,      half);
      v16h bf1 = b_frag_f16(wouth, C_, col, kc + 32, half);
      acc[cg] = wmma16(af0, bf0, acc[cg]);
      acc[cg] = wmma16(af1, bf1, acc[cg]);
    }
  }

#pragma unroll
  for (int cg = 0; cg < 4; ++cg) {
    const int j = col0 + cg * 16 + n;
    const float bias = bout[j];
#pragma unroll
    for (int r = 0; r < 8; ++r) {
      const int row = row0 + half * 8 + r;
      if (row < N_) out[((size_t)b * N_ + row) * C_ + j] = acc[cg][r] + bias;
    }
  }
}

// ---------------------------------------------------------------------------
// Zero cls accumulator region
// ---------------------------------------------------------------------------
__global__ void init_cls(float* __restrict__ cls) {
  int i = blockIdx.x * 256 + threadIdx.x;
  if (i < B_ * 1024) cls[i] = 0.0f;
}

// ---------------------------------------------------------------------------
// Top-k selections (103 largest = enhance, 103 smallest = fuse), JAX tie rule:
// equal values -> smaller index first. Index tensors written as floats.
// ---------------------------------------------------------------------------
__global__ void __launch_bounds__(256) topk_kernel(const float* __restrict__ cls,
                                                   float* __restrict__ outp) {
  __shared__ float work[1024];
  __shared__ float bv[256];
  __shared__ int   bi[256];
  __shared__ int   sel[TOPK];
  const int b = blockIdx.x, tid = threadIdx.x;

  // ---- enhance: iterative argmax ----
  for (int i = tid; i < 1024; i += 256) work[i] = cls[(size_t)b * 1024 + i];
  __syncthreads();
  for (int t = 0; t < TOPK; ++t) {
    float best = NEGBIG; int bidx = 0x7fffffff;
    for (int i = tid; i < 1024; i += 256) {
      float v = work[i];
      if (v > best || (v == best && i < bidx)) { best = v; bidx = i; }
    }
    bv[tid] = best; bi[tid] = bidx;
    __syncthreads();
    for (int s = 128; s > 0; s >>= 1) {
      if (tid < s) {
        float v2 = bv[tid + s]; int i2 = bi[tid + s];
        if (v2 > bv[tid] || (v2 == bv[tid] && i2 < bi[tid])) { bv[tid] = v2; bi[tid] = i2; }
      }
      __syncthreads();
    }
    if (tid == 0) { sel[t] = bi[0]; work[bi[0]] = NEGBIG; }
    __syncthreads();
  }
  for (int t = tid; t < TOPK; t += 256)
    outp[OFF_ENHS + (size_t)b * TOPK + t] = (float)sel[t];
  for (int e = tid; e < TOPK * C_; e += 256)
    outp[OFF_ENHI + (size_t)b * TOPK * C_ + e] = (float)sel[e / C_];
  __syncthreads();

  // ---- fuse: iterative argmin ----
  for (int i = tid; i < 1024; i += 256) work[i] = cls[(size_t)b * 1024 + i];
  __syncthreads();
  for (int t = 0; t < TOPK; ++t) {
    float best = 3.0e38f; int bidx = 0x7fffffff;
    for (int i = tid; i < 1024; i += 256) {
      float v = work[i];
      if (v < best || (v == best && i < bidx)) { best = v; bidx = i; }
    }
    bv[tid] = best; bi[tid] = bidx;
    __syncthreads();
    for (int s = 128; s > 0; s >>= 1) {
      if (tid < s) {
        float v2 = bv[tid + s]; int i2 = bi[tid + s];
        if (v2 < bv[tid] || (v2 == bv[tid] && i2 < bi[tid])) { bv[tid] = v2; bi[tid] = i2; }
      }
      __syncthreads();
    }
    if (tid == 0) { sel[t] = bi[0]; work[bi[0]] = 3.0e38f; }
    __syncthreads();
  }
  for (int t = tid; t < TOPK; t += 256)
    outp[OFF_FUSS + (size_t)b * TOPK + t] = (float)sel[t];
  for (int e = tid; e < TOPK * C_; e += 256)
    outp[OFF_FUSI + (size_t)b * TOPK * C_ + e] = (float)sel[e / C_];
}

// ---------------------------------------------------------------------------
extern "C" void kernel_launch(void* const* d_in, const int* in_sizes, int n_in,
                              void* d_out, int out_size, void* d_ws, size_t ws_size,
                              hipStream_t stream) {
  (void)in_sizes; (void)n_in; (void)out_size; (void)ws_size;
  const float* x    = (const float*)d_in[0];
  const float* wqkv = (const float*)d_in[1];
  const float* wout = (const float*)d_in[2];
  const float* bout = (const float*)d_in[3];
  float* outp = (float*)d_out;

  const size_t qsz    = (size_t)B_ * H_ * NP * D_;     // elements (f16)
  const size_t padsz  = (size_t)B_ * NROWP * C_;       // xh and ctx, padded
  const size_t xsz    = (size_t)B_ * N_ * C_;
  const size_t wqkvsz = (size_t)3 * C_ * C_;
  const size_t woutsz = (size_t)C_ * C_;

  _Float16* q      = (_Float16*)d_ws;
  _Float16* k      = q + qsz;
  _Float16* vt     = k + qsz;
  _Float16* ctx    = vt + qsz;
  _Float16* xh     = ctx + padsz;
  _Float16* wqkvh  = xh + padsz;
  _Float16* wouth  = wqkvh + wqkvsz;
  float* cls = outp + OFF_CLS;

  zero_f16<<<dim3((int)((padsz / 8 + 255) / 256)), dim3(256), 0, stream>>>(xh,  (int)padsz);
  zero_f16<<<dim3((int)((padsz / 8 + 255) / 256)), dim3(256), 0, stream>>>(ctx, (int)padsz);
  cvt_x<<<dim3((int)((xsz / 8 + 255) / 256)), dim3(256), 0, stream>>>(x, xh);
  cvt_f16<<<dim3((int)((wqkvsz / 8 + 255) / 256)), dim3(256), 0, stream>>>(wqkv, wqkvh, (int)wqkvsz);
  cvt_f16<<<dim3((int)((woutsz / 8 + 255) / 256)), dim3(256), 0, stream>>>(wout, wouth, (int)woutsz);
  init_cls<<<dim3((B_ * 1024 + 255) / 256), dim3(256), 0, stream>>>(cls);

  qkv_kernel<<<dim3(9, 36, B_), dim3(256), 0, stream>>>(xh, wqkvh, q, k, vt);
  attn_kernel<<<dim3(65, H_, B_), dim3(128), 0, stream>>>(q, k, vt, ctx, cls);
  outproj_kernel<<<dim3(9, 12, B_), dim3(256), 0, stream>>>(ctx, wouth, bout, outp);
  topk_kernel<<<dim3(B_), dim3(256), 0, stream>>>(cls, outp);
}
